// MultiHeadCrossAttention_76587856822280
// MI455X (gfx1250) — matrix operand order repaired and checked by reading, not compiled, not run on hardware
//
#include <hip/hip_runtime.h>
#include <hip/hip_bf16.h>
#include <cstdint>

// ---------------------------------------------------------------------------
// MHA w/ RoPE for MI455X (gfx1250): all matmuls via v_wmma_f32_16x16x32_bf16,
// tiles staged to LDS with gfx1250 async global->LDS copies (ASYNCcnt),
// double-buffered. B=2, S=2048, D=2048, H=16, hd=128. fp32 in/out, bf16 core.
// ---------------------------------------------------------------------------

#define SEQ  2048
#define DIM  2048
#define NH   16
#define HDIM 128
#define NB   2
#define BHN  (NB * NH)

typedef __bf16 bf16_t;
typedef __attribute__((ext_vector_type(16))) __bf16 v16bf;
typedef __attribute__((ext_vector_type(8)))  float  v8f;

// ---- async global->LDS (gfx1250 GLOBAL_LOAD_ASYNC_TO_LDS_B128) -------------
#if defined(__AMDGCN__) && __has_builtin(__builtin_amdgcn_global_load_async_to_lds_b128) && __has_builtin(__builtin_amdgcn_s_wait_asynccnt)
#define USE_ASYNC 1
#else
#define USE_ASYNC 0
#endif

#if USE_ASYNC
typedef int v4i_vec __attribute__((__vector_size__(4 * sizeof(int))));
typedef __attribute__((address_space(1))) v4i_vec gv4i;   // global (device) ptr
typedef __attribute__((address_space(3))) v4i_vec sv4i;   // LDS ptr
__device__ __forceinline__ void g2s_b128(const void* g, void* s) {
    // LDS operand wants the wave-relative LDS address = low 32 bits of the
    // generic pointer (flat LDS aperture truncation, ISA 10.2).
    __builtin_amdgcn_global_load_async_to_lds_b128(
        (gv4i*)(uintptr_t)g, (sv4i*)(uint32_t)(uintptr_t)s, 0, 0);
}
#define ASYNC_WAIT(N) __builtin_amdgcn_s_wait_asynccnt(N)
#else
__device__ __forceinline__ void g2s_b128(const void* g, void* s) {
    *(uint4*)s = *(const uint4*)g;
}
#define ASYNC_WAIT(N) do {} while (0)
#endif

// Load two 8-halfword (16B) chunks into a WMMA fragment register set.
__device__ __forceinline__ v16bf load2x8(const bf16_t* p0, const bf16_t* p1) {
    union { v16bf v; uint4 q[2]; } u;
    u.q[0] = *reinterpret_cast<const uint4*>(p0);
    u.q[1] = *reinterpret_cast<const uint4*>(p1);
    return u.v;
}

__device__ __forceinline__ v8f wmma_bf16(v16bf a, v16bf b, v8f c) {
    // (neg_a, A, neg_b, B, c_mod, C, reuse_a, reuse_b)
    return __builtin_amdgcn_wmma_f32_16x16x32_bf16(false, a, false, b,
                                                   (short)0, c, false, false);
}

// ---------------------------------------------------------------------------
// fp32 -> bf16 bulk convert (vectorized x4)
// ---------------------------------------------------------------------------
__global__ __launch_bounds__(256)
void cvt_f32_bf16(const float* __restrict__ in, bf16_t* __restrict__ out, int n) {
    int i = (blockIdx.x * 256 + threadIdx.x) * 4;
    if (i + 3 < n) {
        float4 v = *reinterpret_cast<const float4*>(in + i);
        out[i + 0] = (bf16_t)v.x;
        out[i + 1] = (bf16_t)v.y;
        out[i + 2] = (bf16_t)v.z;
        out[i + 3] = (bf16_t)v.w;
    }
}

// ---------------------------------------------------------------------------
// C = A @ W^T.  A:[M,K] bf16 row-major, W:[N,K] bf16 row-major (torch Linear).
// Block = 256 thr (8 waves), block tile 128x128, wave tile 32x64 (2x4 WMMA).
// K-loop: 128x32 A and W tiles double-buffered in LDS via async copies.
// MODE 0: bf16 out [B,H,S,hd] | MODE 1: bf16 out [B,H,hd,S] | MODE 2: f32 [M,N]
// ---------------------------------------------------------------------------
#define GPAD 40   // row stride in halves (80B): conflict-free, 16B-aligned

template <int MODE>
__global__ __launch_bounds__(256)
void gemm_xwt(const bf16_t* __restrict__ A, const bf16_t* __restrict__ W,
              void* __restrict__ outp, int M, int N, int K) {
    const int t    = threadIdx.x;
    const int lane = t & 31;
    const int wave = t >> 5;
    const int lh = lane & 15;
    const int lg = lane >> 4;
    const int mB = blockIdx.x * 128;
    const int nB = blockIdx.y * 128;
    const int m0 = (wave >> 1) * 32;
    const int n0 = (wave & 1) * 64;

    __shared__ __align__(16) bf16_t At[2][128][GPAD];
    __shared__ __align__(16) bf16_t Bt[2][128][GPAD];

    v8f acc[2][4];
    v8f zero = {};
#pragma unroll
    for (int i = 0; i < 2; ++i)
#pragma unroll
        for (int j = 0; j < 4; ++j) acc[i][j] = zero;

    const int NT = K / 32;
    // stage one 128x32 tile pair: 512 16B chunks each -> 2 per thread per tile
    auto stage = [&](int kt, int bf) {
#pragma unroll
        for (int r = 0; r < 2; ++r) {
            int c = t + r * 256;
            int row = c >> 2, cc = c & 3;
            g2s_b128(A + (size_t)(mB + row) * K + kt * 32 + cc * 8, &At[bf][row][cc * 8]);
            g2s_b128(W + (size_t)(nB + row) * K + kt * 32 + cc * 8, &Bt[bf][row][cc * 8]);
        }
    };
    stage(0, 0);

    for (int kt = 0; kt < NT; ++kt) {
        const int cur = kt & 1;
        if (kt + 1 < NT) { stage(kt + 1, cur ^ 1); ASYNC_WAIT(4); }
        else            { ASYNC_WAIT(0); }
        __syncthreads();   // tile kt visible from all waves

        v16bf af[2];
#pragma unroll
        for (int ms = 0; ms < 2; ++ms) {
            const bf16_t* pa = &At[cur][m0 + ms * 16 + lh][lg * 8];
            af[ms] = load2x8(pa, pa + 16);
        }
#pragma unroll
        for (int ns = 0; ns < 4; ++ns) {
            const bf16_t* pb = &Bt[cur][n0 + ns * 16 + lh][lg * 8];
            v16bf bfr = load2x8(pb, pb + 16);
#pragma unroll
            for (int ms = 0; ms < 2; ++ms)
                acc[ms][ns] = wmma_bf16(af[ms], bfr, acc[ms][ns]);
        }
        __syncthreads();   // reads done before buffer reuse
    }

    // Epilogue: C/D layout -> element (m = mB+m0+ms*16+lg*8+j, n = nB+n0+ns*16+lh)
#pragma unroll
    for (int ms = 0; ms < 2; ++ms)
#pragma unroll
        for (int ns = 0; ns < 4; ++ns)
#pragma unroll
            for (int j = 0; j < 8; ++j) {
                int m = mB + m0 + ms * 16 + lg * 8 + j;
                int n = nB + n0 + ns * 16 + lh;
                float c = acc[ms][ns][j];
                if constexpr (MODE == 0) {
                    int b = m / SEQ, s = m % SEQ, h = n / HDIM, d = n % HDIM;
                    ((bf16_t*)outp)[(((size_t)b * NH + h) * SEQ + s) * HDIM + d] = (bf16_t)c;
                } else if constexpr (MODE == 1) {
                    int b = m / SEQ, s = m % SEQ, h = n / HDIM, d = n % HDIM;
                    ((bf16_t*)outp)[(((size_t)b * NH + h) * HDIM + d) * SEQ + s] = (bf16_t)c;
                } else {
                    ((float*)outp)[(size_t)m * N + n] = c;
                }
            }
}

// ---------------------------------------------------------------------------
// RoPE in-place on [B,H,S,hd] bf16; pairs (d, d+64); folds 1/sqrt(hd) into Q.
// ---------------------------------------------------------------------------
__global__ __launch_bounds__(256)
void rope_kernel(bf16_t* __restrict__ X, float scale) {
    int idx = blockIdx.x * 256 + threadIdx.x;
    int d  = idx & 63;
    int s  = (idx >> 6) & (SEQ - 1);
    int bh = idx >> 17;                       // 64 * 2048 = 2^17
    if (bh >= BHN) return;
    float inv = __powf(10000.0f, -(float)d * (1.0f / 64.0f));
    float ang = (float)s * inv;
    float c = __cosf(ang), sn = __sinf(ang);
    size_t base = ((size_t)bh * SEQ + s) * HDIM;
    float x0 = (float)X[base + d];
    float x1 = (float)X[base + d + 64];
    X[base + d]      = (bf16_t)((x0 * c - x1 * sn) * scale);
    X[base + d + 64] = (bf16_t)((x1 * c + x0 * sn) * scale);
}

// ---------------------------------------------------------------------------
// Flash attention per (b,h). Block = 8 waves x 16 q-rows = 128 q / block.
// K tile (32x128) and Vt tile (128x32) staged once per block into LDS with
// async copies, double-buffered; all 8 waves share them via ds_load_b128.
// Q:[bh,S,hd] (pre-scaled), K:[bh,S,hd], Vt:[bh,hd,S] bf16. mask:[B,S,S] i32.
// Out: bf16 [B,S,H,hd] (A-matrix of final Wo GEMM).
// ---------------------------------------------------------------------------
#define KPAD 136  // 272B row stride: conflict-free, 16B-aligned
#define VPAD 40   // 80B  row stride: conflict-free, 16B-aligned

__global__ __launch_bounds__(256)
void flash_attn(const bf16_t* __restrict__ Qh, const bf16_t* __restrict__ Kh,
                const bf16_t* __restrict__ Vt, const int* __restrict__ mask,
                bf16_t* __restrict__ Out) {
    const int t    = threadIdx.x;
    const int lane = t & 31;
    const int wave = t >> 5;
    const int lh = lane & 15;
    const int lg = lane >> 4;
    const int bh = blockIdx.y;
    const int b  = bh / NH;
    const int h  = bh % NH;
    const int q0 = blockIdx.x * 128 + wave * 16;

    const bf16_t* Qp = Qh + (size_t)bh * SEQ * HDIM;
    const bf16_t* Kp = Kh + (size_t)bh * SEQ * HDIM;
    const bf16_t* Vp = Vt + (size_t)bh * HDIM * SEQ;

    __shared__ __align__(16) bf16_t Kt [2][32][KPAD];   // [kv_local][hdim]
    __shared__ __align__(16) bf16_t Vtl[2][128][VPAD];  // [d][kv_local]
    __shared__ __align__(16) bf16_t plds[8][16][48];    // per-wave P tile

    // Resident Q fragments covering hd=128 (4 x K=32)
    v16bf qf[4];
#pragma unroll
    for (int ks = 0; ks < 4; ++ks) {
        const bf16_t* p = Qp + (size_t)(q0 + lh) * HDIM + ks * 32 + lg * 8;
        qf[ks] = load2x8(p, p + 16);
    }

    v8f o[8];
    v8f zero = {};
#pragma unroll
    for (int dt = 0; dt < 8; ++dt) o[dt] = zero;
    float mrow[8], lrow[8];
#pragma unroll
    for (int j = 0; j < 8; ++j) { mrow[j] = -3.0e38f; lrow[j] = 0.0f; }

    // stage one 32-wide kv tile: K 512 chunks + V 512 chunks -> 4 per thread
    auto stage = [&](int kv, int bf) {
#pragma unroll
        for (int r = 0; r < 2; ++r) {
            int c = t + r * 256;
            int krow = c >> 4, kcol = c & 15;
            g2s_b128(Kp + (size_t)(kv + krow) * HDIM + kcol * 8, &Kt[bf][krow][kcol * 8]);
            int vd = c >> 2, vc = c & 3;
            g2s_b128(Vp + (size_t)vd * SEQ + kv + vc * 8, &Vtl[bf][vd][vc * 8]);
        }
    };
    stage(0, 0);

    const int NT = SEQ / 32;
    for (int it = 0; it < NT; ++it) {
        const int cur = it & 1;
        const int kv  = it * 32;
        if (it + 1 < NT) { stage(kv + 32, cur ^ 1); ASYNC_WAIT(4); }
        else             { ASYNC_WAIT(0); }
        __syncthreads();   // (B1) K/V tile `it` visible; prior plds reads done

        // ---- scores S = Q @ K^T (16x32) from LDS K tile ----
        v8f sc[2];
        sc[0] = zero; sc[1] = zero;
#pragma unroll
        for (int ks = 0; ks < 4; ++ks)
#pragma unroll
            for (int ns = 0; ns < 2; ++ns) {
                const bf16_t* p = &Kt[cur][ns * 16 + lh][ks * 32 + lg * 8];
                v16bf kf = load2x8(p, p + 16);
                sc[ns] = wmma_bf16(qf[ks], kf, sc[ns]);
            }

        // ---- additive mask (reference: where(mask==0, -inf)) ----
#pragma unroll
        for (int ns = 0; ns < 2; ++ns)
#pragma unroll
            for (int j = 0; j < 8; ++j) {
                int q  = q0 + lg * 8 + j;
                int kk = kv + ns * 16 + lh;
                if (mask[((size_t)b * SEQ + q) * SEQ + kk] == 0)
                    sc[ns][j] = -3.0e38f;
            }

        // ---- streaming softmax: row max over 32 cols (16-lane butterflies) ----
        float corr[8];
#pragma unroll
        for (int j = 0; j < 8; ++j) {
            float tm = fmaxf(sc[0][j], sc[1][j]);
#pragma unroll
            for (int off = 8; off >= 1; off >>= 1)
                tm = fmaxf(tm, __shfl_xor(tm, off, 32));
            float mn = fmaxf(mrow[j], tm);
            corr[j]  = __expf(mrow[j] - mn);
            mrow[j]  = mn;
        }
#pragma unroll
        for (int j = 0; j < 8; ++j) {
            float p0 = __expf(sc[0][j] - mrow[j]);
            float p1 = __expf(sc[1][j] - mrow[j]);
            plds[wave][lg * 8 + j][lh]      = (bf16_t)p0;
            plds[wave][lg * 8 + j][16 + lh] = (bf16_t)p1;
            float rs = p0 + p1;
#pragma unroll
            for (int off = 8; off >= 1; off >>= 1)
                rs += __shfl_xor(rs, off, 32);
            lrow[j] = lrow[j] * corr[j] + rs;
        }
#pragma unroll
        for (int dt = 0; dt < 8; ++dt)
#pragma unroll
            for (int j = 0; j < 8; ++j) o[dt][j] *= corr[j];
        __syncthreads();   // (B2) P tile visible

        // ---- P (A-layout via LDS) x V (LDS B-operand) ----
        const bf16_t* pr = &plds[wave][lh][0];
        v16bf pf = load2x8(pr + lg * 8, pr + 16 + lg * 8);
#pragma unroll
        for (int dt = 0; dt < 8; ++dt) {
            const bf16_t* p = &Vtl[cur][dt * 16 + lh][lg * 8];
            v16bf vf = load2x8(p, p + 16);
            o[dt] = wmma_bf16(pf, vf, o[dt]);
        }
        __syncthreads();   // (B3) V reads done before buffer reuse
    }

    // ---- normalize + scatter to [B,S,H,hd] ----
#pragma unroll
    for (int dt = 0; dt < 8; ++dt)
#pragma unroll
        for (int j = 0; j < 8; ++j) {
            int q = q0 + lg * 8 + j;
            int d = dt * 16 + lh;
            float v = o[dt][j] / lrow[j];
            Out[(((size_t)b * SEQ + q) * NH + h) * HDIM + d] = (bf16_t)v;
        }
}

// ---------------------------------------------------------------------------
// Launch
// ---------------------------------------------------------------------------
extern "C" void kernel_launch(void* const* d_in, const int* in_sizes, int n_in,
                              void* d_out, int out_size, void* d_ws, size_t ws_size,
                              hipStream_t stream) {
    const float* query = (const float*)d_in[0];
    const float* key   = (const float*)d_in[1];
    const float* value = (const float*)d_in[2];
    const int*   amask = (const int*)  d_in[3];
    const float* wq    = (const float*)d_in[4];
    const float* wk    = (const float*)d_in[5];
    const float* wv    = (const float*)d_in[6];
    const float* wo    = (const float*)d_in[7];
    float* out = (float*)d_out;

    const size_t WSZ = (size_t)DIM * DIM;        // 4M elems (weight)
    const size_t XSZ = (size_t)NB * SEQ * DIM;   // 8M elems (activation)
    bf16_t* ws  = (bf16_t*)d_ws;                 // total 72M bf16 = 144 MB
    bf16_t* wqb = ws;
    bf16_t* wkb = wqb + WSZ;
    bf16_t* wvb = wkb + WSZ;
    bf16_t* wob = wvb + WSZ;
    bf16_t* xq  = wob + WSZ;
    bf16_t* xk  = xq + XSZ;
    bf16_t* xv  = xk + XSZ;
    bf16_t* qh  = xv + XSZ;   // [B,H,S,hd]
    bf16_t* kh  = qh + XSZ;   // [B,H,S,hd]
    bf16_t* vt  = kh + XSZ;   // [B,H,hd,S]
    bf16_t* ao  = vt + XSZ;   // [B,S,H*hd]

    int wb = (int)(WSZ / 4 / 256);   // 4096 blocks
    int xb = (int)(XSZ / 4 / 256);   // 8192 blocks
    cvt_f32_bf16<<<wb, 256, 0, stream>>>(wq, wqb, (int)WSZ);
    cvt_f32_bf16<<<wb, 256, 0, stream>>>(wk, wkb, (int)WSZ);
    cvt_f32_bf16<<<wb, 256, 0, stream>>>(wv, wvb, (int)WSZ);
    cvt_f32_bf16<<<wb, 256, 0, stream>>>(wo, wob, (int)WSZ);
    cvt_f32_bf16<<<xb, 256, 0, stream>>>(query, xq, (int)XSZ);
    cvt_f32_bf16<<<xb, 256, 0, stream>>>(key,   xk, (int)XSZ);
    cvt_f32_bf16<<<xb, 256, 0, stream>>>(value, xv, (int)XSZ);

    dim3 gg(NB * SEQ / 128, DIM / 128);   // (32, 16)
    gemm_xwt<0><<<gg, 256, 0, stream>>>(xq, wqb, qh, NB * SEQ, DIM, DIM);
    gemm_xwt<0><<<gg, 256, 0, stream>>>(xk, wkb, kh, NB * SEQ, DIM, DIM);
    gemm_xwt<1><<<gg, 256, 0, stream>>>(xv, wvb, vt, NB * SEQ, DIM, DIM);

    int rb = (BHN * SEQ * 64) / 256;      // 16384 blocks
    rope_kernel<<<rb, 256, 0, stream>>>(qh, 0.088388347648318447f);
    rope_kernel<<<rb, 256, 0, stream>>>(kh, 1.0f);

    flash_attn<<<dim3(SEQ / 128, BHN), 256, 0, stream>>>(qh, kh, vt, amask, ao);

    gemm_xwt<2><<<gg, 256, 0, stream>>>(ao, wob, out, NB * SEQ, DIM, DIM);
}
